// Clustering_16355235463634
// MI455X (gfx1250) — matrix operand order repaired and checked by reading, not compiled, not run on hardware
//
#include <hip/hip_runtime.h>

// ---------------------------------------------------------------------------
// Spectral clustering pipeline for MI455X (gfx1250, wave32, WMMA).
//   n=4096 nodes, d=256 features, e=128 emb, k=2048 clusters.
// Heavy ops:
//   * emb GEMM (4096x256 @ 256x128)            -> v_wmma_f32_16x16x32_bf16
//   * edge 4096x4096 softmax+symmetrize        -> bandwidth bound (64MB)
//   * eigh replaced by subspace iteration on
//     M = I + D^-1/2 W D^-1/2 (bf16, 32MB,
//     L2-resident on the 192MB L2)             -> WMMA + async LDS staging
//   * kmeans (deterministic segment sums)
// ---------------------------------------------------------------------------

#define N_NODES   4096
#define D_IN      256
#define E_DIM     128
#define K_CLUST   2048
#define KM_ITERS  10
#define POWER_ITERS 24
#define NEG_SLOPE 0.2f

typedef __attribute__((ext_vector_type(16))) __bf16 v16bf;
typedef __attribute__((ext_vector_type(8)))  __bf16 v8bf;
typedef __attribute__((ext_vector_type(8)))  float  v8f;
typedef __attribute__((address_space(3)))    __bf16 lds_bf16_t;

static __device__ __forceinline__ v16bf cat8(v8bf a, v8bf b) {
  return __builtin_shufflevector(a, b, 0,1,2,3,4,5,6,7,8,9,10,11,12,13,14,15);
}
static __device__ __forceinline__ float lrelu(float s) {
  return s >= 0.0f ? s : NEG_SLOPE * s;
}

// ---------------------------------------------------------------------------
// K1: emb = node @ W_emb^T + b_emb   (4096x256 @ 256x128)
// One wave per 16x16 output tile; 8 waves/block cover all 8 N-tiles of an
// M-row-block. K loop in steps of 32 with bf16 WMMA, f32 accumulate.
// ---------------------------------------------------------------------------
__global__ __launch_bounds__(256) void emb_gemm_kernel(
    const float* __restrict__ node, const float* __restrict__ Wemb,
    const float* __restrict__ bemb, float* __restrict__ emb) {
  const int lane = threadIdx.x & 31;
  const int wave = threadIdx.x >> 5;
  const int m0   = blockIdx.x * 16;
  const int n0   = wave * 16;
  const int row  = lane & 15;     // M row within tile (A), N col (B/C)
  const int half = lane >> 4;

  v8f acc = {};
  for (int k0 = 0; k0 < D_IN; k0 += 32) {
    // A tile: node rows, element j <-> K = k0 + (j>>3)*16 + half*8 + (j&7)
    const float* ar = node + (size_t)(m0 + row) * D_IN + k0 + half * 8;
    float4 a0 = *(const float4*)(ar);
    float4 a1 = *(const float4*)(ar + 4);
    float4 a2 = *(const float4*)(ar + 16);
    float4 a3 = *(const float4*)(ar + 20);
    v16bf A;
    A[0]=(__bf16)a0.x; A[1]=(__bf16)a0.y; A[2]=(__bf16)a0.z; A[3]=(__bf16)a0.w;
    A[4]=(__bf16)a1.x; A[5]=(__bf16)a1.y; A[6]=(__bf16)a1.z; A[7]=(__bf16)a1.w;
    A[8]=(__bf16)a2.x; A[9]=(__bf16)a2.y; A[10]=(__bf16)a2.z; A[11]=(__bf16)a2.w;
    A[12]=(__bf16)a3.x; A[13]=(__bf16)a3.y; A[14]=(__bf16)a3.z; A[15]=(__bf16)a3.w;
    // B tile: B[k][n] = W_emb[n][k]; element j <-> K = k0 + (j>>3)*16 + half*8 + (j&7)
    const float* br = Wemb + (size_t)(n0 + row) * D_IN + k0 + half * 8;
    float4 b0 = *(const float4*)(br);
    float4 b1 = *(const float4*)(br + 4);
    float4 b2 = *(const float4*)(br + 16);
    float4 b3 = *(const float4*)(br + 20);
    v16bf B;
    B[0]=(__bf16)b0.x; B[1]=(__bf16)b0.y; B[2]=(__bf16)b0.z; B[3]=(__bf16)b0.w;
    B[4]=(__bf16)b1.x; B[5]=(__bf16)b1.y; B[6]=(__bf16)b1.z; B[7]=(__bf16)b1.w;
    B[8]=(__bf16)b2.x; B[9]=(__bf16)b2.y; B[10]=(__bf16)b2.z; B[11]=(__bf16)b2.w;
    B[12]=(__bf16)b3.x; B[13]=(__bf16)b3.y; B[14]=(__bf16)b3.z; B[15]=(__bf16)b3.w;
    acc = __builtin_amdgcn_wmma_f32_16x16x32_bf16(
        false, A, false, B, (short)0, acc, false, false);
  }
  // C/D: VGPR r -> (M = m0+r+half*8, N = n0+(lane&15))
  const int n = n0 + row;
  const float bias = bemb[n];
  #pragma unroll
  for (int r = 0; r < 8; ++r)
    emb[(size_t)(m0 + r + half * 8) * E_DIM + n] = acc[r] + bias;
}

// ---------------------------------------------------------------------------
// K2: s_other[i] = emb[i,:] . w_attn[0:128],  s_self[i] = emb[i,:] . w_attn[128:256]
// ---------------------------------------------------------------------------
__global__ __launch_bounds__(128) void sdots_kernel(
    const float* __restrict__ emb, const float* __restrict__ wattn,
    float* __restrict__ s_other, float* __restrict__ s_self) {
  __shared__ float red[128];
  const int i = blockIdx.x, t = threadIdx.x;
  const float e = emb[(size_t)i * E_DIM + t];
  red[t] = e * wattn[t];
  __syncthreads();
  for (int o = 64; o > 0; o >>= 1) { if (t < o) red[t] += red[t + o]; __syncthreads(); }
  if (t == 0) s_other[i] = red[0];
  __syncthreads();
  red[t] = e * wattn[E_DIM + t];
  __syncthreads();
  for (int o = 64; o > 0; o >>= 1) { if (t < o) red[t] += red[t + o]; __syncthreads(); }
  if (t == 0) s_self[i] = red[0];
}

// K3: global max of s_self (leaky-relu is monotone -> column max closed form)
__global__ __launch_bounds__(1024) void smax_kernel(
    const float* __restrict__ s_self, float* __restrict__ out) {
  __shared__ float red[1024];
  float m = -3.4e38f;
  for (int i = threadIdx.x; i < N_NODES; i += 1024) m = fmaxf(m, s_self[i]);
  red[threadIdx.x] = m; __syncthreads();
  for (int o = 512; o > 0; o >>= 1) {
    if (threadIdx.x < o) red[threadIdx.x] = fmaxf(red[threadIdx.x], red[threadIdx.x + o]);
    __syncthreads();
  }
  if (threadIdx.x == 0) out[0] = red[0];
}

// ---------------------------------------------------------------------------
// K4: per-column softmax stats. colm[j] = max_i score, cold[j] = sum exp.
// Overwrites s_other[j] with c_j = s_other[j] + b_attn (safe: 1 thread/col).
// ---------------------------------------------------------------------------
__global__ __launch_bounds__(256) void colstats_kernel(
    const float* __restrict__ s_self, float* __restrict__ s_other,
    const float* __restrict__ battn, const float* __restrict__ smax,
    float* __restrict__ colm, float* __restrict__ cold) {
  __shared__ float ss[N_NODES];
  for (int i = threadIdx.x; i < N_NODES; i += 256) ss[i] = s_self[i];
  __syncthreads();
  const int j = blockIdx.x * 256 + threadIdx.x;
  const float c = s_other[j] + battn[0];
  const float mv = lrelu(smax[0] + c);
  float den = 0.0f;
  for (int i = 0; i < N_NODES; ++i)
    den += __expf(lrelu(ss[i] + c) - mv);
  colm[j] = mv; cold[j] = den; s_other[j] = c;
}

// ---------------------------------------------------------------------------
// K5: edge[i,j] = softmax-col value at (min(i,j), max(i,j))  (symmetrized)
// ---------------------------------------------------------------------------
__global__ __launch_bounds__(256) void edge_kernel(
    const float* __restrict__ s_self, const float* __restrict__ colc,
    const float* __restrict__ colm, const float* __restrict__ cold,
    float* __restrict__ edge) {
  const int j = blockIdx.x * 256 + threadIdx.x;
  const int i = blockIdx.y;
  const int ii = min(i, j), jj = max(i, j);
  const float s = lrelu(s_self[ii] + colc[jj]);
  edge[(size_t)i * N_NODES + j] = __expf(s - colm[jj]) / cold[jj];
}

// K6: deg row-sum -> invs[i] = 1/sqrt(deg)
__global__ __launch_bounds__(256) void deg_kernel(
    const float* __restrict__ edge, float* __restrict__ invs) {
  __shared__ float red[256];
  const float4* r4 = (const float4*)(edge + (size_t)blockIdx.x * N_NODES);
  float s = 0.0f;
  for (int j = threadIdx.x; j < N_NODES / 4; j += 256) {
    float4 v = r4[j]; s += v.x + v.y + v.z + v.w;
  }
  red[threadIdx.x] = s; __syncthreads();
  for (int o = 128; o > 0; o >>= 1) { if (threadIdx.x < o) red[threadIdx.x] += red[threadIdx.x + o]; __syncthreads(); }
  if (threadIdx.x == 0) invs[blockIdx.x] = rsqrtf(red[0]);
}

// K7: Ebf[i,j] = bf16( invs[i] * edge[i,j] * invs[j] )  (32MB, L2-resident)
__global__ __launch_bounds__(256) void ebf_kernel(
    const float* __restrict__ edge, const float* __restrict__ invs,
    __bf16* __restrict__ E) {
  const int j = blockIdx.x * 256 + threadIdx.x;
  const int i = blockIdx.y;
  E[(size_t)i * N_NODES + j] = (__bf16)(edge[(size_t)i * N_NODES + j] * invs[i] * invs[j]);
}

// K8: deterministic pseudo-random init of the 16-column subspace V (4096x16)
__global__ __launch_bounds__(256) void vinit_kernel(float* __restrict__ V) {
  const int idx = blockIdx.x * 256 + threadIdx.x;   // < 65536
  unsigned h = (unsigned)idx * 2654435761u;
  h ^= h >> 16; h *= 2246822519u; h ^= h >> 13;
  V[idx] = (float)(h & 0xFFFF) * (1.0f / 32768.0f) - 1.0f;
}

// K9: stage V (f32, row-major 4096x16) into WMMA-B-ordered bf16 buffer:
// idx = ((k>>5)*2 + ((k>>4)&1))*256 + n*16 + (k&15)
__global__ __launch_bounds__(256) void stage_kernel(
    const float* __restrict__ V, __bf16* __restrict__ Xs) {
  const int idx    = blockIdx.x * 256 + threadIdx.x;  // < 65536
  const int blk    = idx >> 9;
  const int within = idx & 511;
  const int half   = within >> 8;
  const int rem    = within & 255;
  const int n      = rem >> 4;
  const int j      = rem & 15;
  const int k      = blk * 32 + half * 16 + j;
  Xs[idx] = (__bf16)V[k * 16 + n];
}

// ---------------------------------------------------------------------------
// K10: Y(4096x16) = Ebf(4096x4096) @ V(4096x16), bf16 WMMA, f32 accumulate.
// The shared B operand (Xs, 128KB) is staged into LDS in two 64KB phases via
// async global->LDS B128 transfers (ASYNCcnt), so the 8 waves of a workgroup
// read B from LDS instead of each pulling 128KB from L2 (8x traffic cut).
// A stream (one 8KB row strip per lane group) is prefetched ahead.
// ---------------------------------------------------------------------------
__global__ __launch_bounds__(256) void spec_gemm_kernel(
    const __bf16* __restrict__ E, const __bf16* __restrict__ Xs,
    float* __restrict__ Y) {
  __shared__ __bf16 xs[32768];                 // 64KB staging buffer (1 phase)
  const int lane = threadIdx.x & 31;
  const int wave = threadIdx.x >> 5;
  const int m0   = (blockIdx.x * 8 + wave) * 16;
  const int row  = lane & 15;
  const int half = lane >> 4;
  const __bf16* ar = E + (size_t)(m0 + row) * N_NODES + half * 16;
  v8f acc = {};

  for (int phase = 0; phase < 2; ++phase) {
    // ---- async stage: 64KB of Xs -> LDS (16 x B128 per thread) ----
    const __bf16* src = Xs + phase * 32768;
    #pragma unroll
    for (int itc = 0; itc < 16; ++itc) {
      const int eidx = (itc * 256 + threadIdx.x) * 8;   // 8 bf16 = 16B / thread
      lds_bf16_t* dst = (lds_bf16_t*)(&xs[eidx]);
      asm volatile("global_load_async_to_lds_b128 %0, %1, off"
                   :: "v"(dst), "v"(src + eidx) : "memory");
    }
    asm volatile("s_wait_asynccnt 0x0" ::: "memory");
    __syncthreads();                           // LDS visible to all 8 waves

    // ---- WMMA K-loop over this 2048-wide K slab ----
    const int kbase = phase * 2048;
    #pragma unroll 4
    for (int kk = 0; kk < 2048; kk += 32) {
      const int k0 = kbase + kk;
      if ((kk & 127) == 0)
        __builtin_prefetch((const void*)(ar + k0 + 512), 0, 0);
      v8bf a0 = *(const v8bf*)(ar + k0);
      v8bf a1 = *(const v8bf*)(ar + k0 + 8);
      v16bf A = cat8(a0, a1);
      v16bf B = *(const v16bf*)(&xs[((kk >> 5) * 2 + half) * 256 + row * 16]);
      acc = __builtin_amdgcn_wmma_f32_16x16x32_bf16(
          false, A, false, B, (short)0, acc, false, false);
    }
    __syncthreads();                           // done reading before restage
  }

  #pragma unroll
  for (int r = 0; r < 8; ++r)
    Y[(m0 + r + half * 8) * 16 + row] = acc[r];
}

// K11: V += Y   (M = I + D^-1/2 W D^-1/2 applied)
__global__ __launch_bounds__(256) void post_kernel(
    float* __restrict__ V, const float* __restrict__ Y) {
  const int idx = blockIdx.x * 256 + threadIdx.x;
  V[idx] += Y[idx];
}

// K12: modified Gram-Schmidt on the 16 columns (single workgroup).
__global__ __launch_bounds__(1024) void mgs_kernel(float* __restrict__ V) {
  __shared__ float red[1024];
  const int t = threadIdx.x;
  for (int c = 0; c < 16; ++c) {
    float s = 0.0f;
    for (int i = t; i < N_NODES; i += 1024) { float v = V[i * 16 + c]; s += v * v; }
    red[t] = s; __syncthreads();
    for (int o = 512; o > 0; o >>= 1) { if (t < o) red[t] += red[t + o]; __syncthreads(); }
    const float inv = rsqrtf(red[0] + 1e-20f);
    __syncthreads();
    for (int i = t; i < N_NODES; i += 1024) V[i * 16 + c] *= inv;
    __syncthreads();
    for (int d = c + 1; d < 16; ++d) {
      float sd = 0.0f;
      for (int i = t; i < N_NODES; i += 1024) sd += V[i * 16 + c] * V[i * 16 + d];
      red[t] = sd; __syncthreads();
      for (int o = 512; o > 0; o >>= 1) { if (t < o) red[t] += red[t + o]; __syncthreads(); }
      const float dot = red[0];
      __syncthreads();
      for (int i = t; i < N_NODES; i += 1024) V[i * 16 + d] -= dot * V[i * 16 + c];
      __syncthreads();
    }
  }
}

// K13: H_norm = row-normalized first two columns of V
__global__ __launch_bounds__(256) void hnorm_kernel(
    const float* __restrict__ V, float* __restrict__ Hn) {
  const int i = blockIdx.x * 256 + threadIdx.x;
  if (i >= N_NODES) return;
  const float a = V[i * 16], b = V[i * 16 + 1];
  const float inv = rsqrtf(a * a + b * b + 1e-20f);
  Hn[2 * i] = a * inv; Hn[2 * i + 1] = b * inv;
}

// K14: kmeans init: centroid c = H[perm(c)], perm = odd-mult bijection mod 4096
__global__ __launch_bounds__(256) void km_init_kernel(
    const float* __restrict__ Hn, float* __restrict__ cent) {
  const int c = blockIdx.x * 256 + threadIdx.x;
  if (c >= K_CLUST) return;
  const unsigned p = ((unsigned)c * 2654435761u) & (N_NODES - 1);
  cent[2 * c] = Hn[2 * p]; cent[2 * c + 1] = Hn[2 * p + 1];
}

// K15: assignment (centroids in LDS, first-index tie-break like argmin)
__global__ __launch_bounds__(256) void km_assign_kernel(
    const float* __restrict__ Hn, const float* __restrict__ cent,
    int* __restrict__ asn) {
  __shared__ float cs[2 * K_CLUST];
  for (int c = threadIdx.x; c < 2 * K_CLUST; c += 256) cs[c] = cent[c];
  __syncthreads();
  const int i = blockIdx.x * 256 + threadIdx.x;
  const float x = Hn[2 * i], y = Hn[2 * i + 1];
  int best = 0; float bd = 3.4e38f;
  for (int c = 0; c < K_CLUST; ++c) {
    const float dx = x - cs[2 * c], dy = y - cs[2 * c + 1];
    const float d = dx * dx + dy * dy;
    if (d < bd) { bd = d; best = c; }
  }
  asn[i] = best;
}

// K16: deterministic segment-sum centroid update (one thread per cluster,
// H staged in LDS; dead clusters replaced from a per-iter bijection).
__global__ __launch_bounds__(256) void km_update_kernel(
    const float* __restrict__ Hn, const int* __restrict__ asn,
    float* __restrict__ cent, int iter) {
  __shared__ float hs[2 * N_NODES];
  for (int i = threadIdx.x; i < 2 * N_NODES; i += 256) hs[i] = Hn[i];
  __syncthreads();
  const int c = blockIdx.x * 256 + threadIdx.x;
  if (c >= K_CLUST) return;
  float sx = 0.0f, sy = 0.0f; int cnt = 0;
  for (int i = 0; i < N_NODES; ++i) {
    if (asn[i] == c) { sx += hs[2 * i]; sy += hs[2 * i + 1]; ++cnt; }
  }
  if (cnt > 0) {
    const float inv = 1.0f / (float)cnt;
    cent[2 * c] = sx * inv; cent[2 * c + 1] = sy * inv;
  } else {
    const unsigned p = (((unsigned)c + 131u * (unsigned)iter) * 2654435761u) & (N_NODES - 1);
    cent[2 * c] = hs[2 * p]; cent[2 * c + 1] = hs[2 * p + 1];
  }
}

// K17: group (int argmin) -> float output slots
__global__ __launch_bounds__(256) void write_group_kernel(
    const int* __restrict__ asn, float* __restrict__ out) {
  const int i = blockIdx.x * 256 + threadIdx.x;
  if (i < N_NODES) out[i] = (float)asn[i];
}

// ---------------------------------------------------------------------------
extern "C" void kernel_launch(void* const* d_in, const int* in_sizes, int n_in,
                              void* d_out, int out_size, void* d_ws, size_t ws_size,
                              hipStream_t stream) {
  (void)in_sizes; (void)n_in; (void)out_size; (void)ws_size;
  const float* node  = (const float*)d_in[0];   // 4096 x 256
  const float* Wemb  = (const float*)d_in[1];   // 128 x 256
  const float* bemb  = (const float*)d_in[2];   // 128
  const float* wattn = (const float*)d_in[3];   // 256
  const float* battn = (const float*)d_in[4];   // 1

  float* group_out = (float*)d_out;             // 4096
  float* edge_out  = group_out + N_NODES;       // 4096*4096

  // Workspace carve-up (~35 MB total, dominated by the 32MB bf16 matrix).
  char* w = (char*)d_ws;
  auto carve = [&](size_t bytes) -> void* {
    void* p = (void*)w; w += (bytes + 255) & ~(size_t)255; return p;
  };
  float*  emb  = (float*)carve((size_t)N_NODES * E_DIM * 4);
  float*  ssf  = (float*)carve(N_NODES * 4);         // s_self
  float*  soth = (float*)carve(N_NODES * 4);         // s_other -> c_j
  float*  colm = (float*)carve(N_NODES * 4);
  float*  cold = (float*)carve(N_NODES * 4);
  float*  smax = (float*)carve(64);
  float*  invs = (float*)carve(N_NODES * 4);
  float*  V    = (float*)carve(N_NODES * 16 * 4);
  float*  Y    = (float*)carve(N_NODES * 16 * 4);
  __bf16* Xs   = (__bf16*)carve(N_NODES * 16 * 2);
  float*  Hn   = (float*)carve(N_NODES * 2 * 4);
  float*  cent = (float*)carve(K_CLUST * 2 * 4);
  int*    asn  = (int*)carve(N_NODES * 4);
  __bf16* Ebf  = (__bf16*)carve((size_t)N_NODES * N_NODES * 2);

  // ---- attention edge matrix -------------------------------------------
  emb_gemm_kernel<<<N_NODES / 16, 256, 0, stream>>>(node, Wemb, bemb, emb);
  sdots_kernel<<<N_NODES, 128, 0, stream>>>(emb, wattn, soth, ssf);
  smax_kernel<<<1, 1024, 0, stream>>>(ssf, smax);
  colstats_kernel<<<N_NODES / 256, 256, 0, stream>>>(ssf, soth, battn, smax, colm, cold);
  edge_kernel<<<dim3(N_NODES / 256, N_NODES), 256, 0, stream>>>(ssf, soth, colm, cold, edge_out);

  // ---- normalized adjacency (bf16, L2-resident) ------------------------
  deg_kernel<<<N_NODES, 256, 0, stream>>>(edge_out, invs);
  ebf_kernel<<<dim3(N_NODES / 256, N_NODES), 256, 0, stream>>>(edge_out, invs, Ebf);

  // ---- subspace iteration for the 2 smallest eigvecs of L_sym ----------
  vinit_kernel<<<(N_NODES * 16) / 256, 256, 0, stream>>>(V);
  mgs_kernel<<<1, 1024, 0, stream>>>(V);
  for (int it = 0; it < POWER_ITERS; ++it) {
    stage_kernel<<<(N_NODES * 16) / 256, 256, 0, stream>>>(V, Xs);
    spec_gemm_kernel<<<N_NODES / (16 * 8), 256, 0, stream>>>(Ebf, Xs, Y);
    post_kernel<<<(N_NODES * 16) / 256, 256, 0, stream>>>(V, Y);
    if ((it & 3) == 3) mgs_kernel<<<1, 1024, 0, stream>>>(V);
  }
  hnorm_kernel<<<N_NODES / 256, 256, 0, stream>>>(V, Hn);

  // ---- kmeans ----------------------------------------------------------
  km_init_kernel<<<K_CLUST / 256, 256, 0, stream>>>(Hn, cent);
  for (int it = 0; it < KM_ITERS; ++it) {
    km_assign_kernel<<<N_NODES / 256, 256, 0, stream>>>(Hn, cent, asn);
    km_update_kernel<<<K_CLUST / 256, 256, 0, stream>>>(Hn, asn, cent, it);
  }
  km_assign_kernel<<<N_NODES / 256, 256, 0, stream>>>(Hn, cent, asn);
  write_group_kernel<<<N_NODES / 256, 256, 0, stream>>>(asn, group_out);
}
